// BiomarkerCombinationFinder_50199577755926
// MI455X (gfx1250) — compile-verified, matrix operand order checked by reading, not run on hardware
//
#include <hip/hip_runtime.h>
#include <hip/hip_bf16.h>

typedef _Float16 f16;
typedef __attribute__((ext_vector_type(16))) _Float16 v16h;
typedef __attribute__((ext_vector_type(8)))  _Float16 v8h;
typedef __attribute__((ext_vector_type(8)))  float    v8f;
typedef __attribute__((ext_vector_type(2)))  float    v2f;

#define B_   128
#define D_   1024
#define H_   256
#define M_   100
#define K_   5
#define H2_  512      // 2H
#define ND_  102400   // M*D
#define NEG_INF (-3.0e38f)

// fp32 WMMA + TDM path if the toolchain exposes the builtins (device pass only;
// host pass sees 0 and parses the proven f16 fallback body instead).
#if defined(__AMDGCN__) && __has_builtin(__builtin_amdgcn_wmma_f32_16x16x4_f32) && \
    __has_builtin(__builtin_amdgcn_tensor_load_to_lds) && \
    __has_builtin(__builtin_amdgcn_s_wait_tensorcnt)
#define USE_F32_WMMA_TDM 1
#else
#define USE_F32_WMMA_TDM 0
#endif

// ---------------- K1: x@gW1 + b -> LN -> ReLU  ([128,1024]->[128,512]) ----
__global__ __launch_bounds__(256)
void gen_layer1(const float* __restrict__ x, const float* __restrict__ W,
                const float* __restrict__ bias, const float* __restrict__ ls,
                const float* __restrict__ lb, float* __restrict__ h1) {
  __shared__ float xs[D_];
  __shared__ float red[256];
  const int b = blockIdx.x, t = threadIdx.x;
  for (int i = t; i < D_; i += 256) xs[i] = x[b * D_ + i];
  __syncthreads();
  float acc[2];
  #pragma unroll
  for (int o = 0; o < 2; ++o) {
    const int n = t + o * 256;
    float s = bias[n];
    for (int k = 0; k < D_; ++k) s = fmaf(xs[k], W[k * H2_ + n], s);
    acc[o] = s;
  }
  red[t] = acc[0] + acc[1];
  __syncthreads();
  for (int s = 128; s > 0; s >>= 1) { if (t < s) red[t] += red[t + s]; __syncthreads(); }
  const float mean = red[0] * (1.0f / H2_);
  __syncthreads();
  const float d0 = acc[0] - mean, d1 = acc[1] - mean;
  red[t] = d0 * d0 + d1 * d1;
  __syncthreads();
  for (int s = 128; s > 0; s >>= 1) { if (t < s) red[t] += red[t + s]; __syncthreads(); }
  const float rstd = rsqrtf(red[0] * (1.0f / H2_) + 1e-5f);
  #pragma unroll
  for (int o = 0; o < 2; ++o) {
    const int n = t + o * 256;
    const float v = (acc[o] - mean) * rstd * ls[n] + lb[n];
    h1[b * H2_ + n] = fmaxf(v, 0.0f);
  }
}

// -------- K2: h1@gW2 + b -> LN -> ReLU -> f32 and f16 ([128,512]->[128,256])
__global__ __launch_bounds__(256)
void gen_layer2(const float* __restrict__ h1, const float* __restrict__ W,
                const float* __restrict__ bias, const float* __restrict__ ls,
                const float* __restrict__ lb, float* __restrict__ h2,
                f16* __restrict__ h2h) {
  __shared__ float xs[H2_];
  __shared__ float red[256];
  const int b = blockIdx.x, t = threadIdx.x;
  for (int i = t; i < H2_; i += 256) xs[i] = h1[b * H2_ + i];
  __syncthreads();
  float s = bias[t];
  for (int k = 0; k < H2_; ++k) s = fmaf(xs[k], W[k * H_ + t], s);
  red[t] = s; __syncthreads();
  for (int st = 128; st > 0; st >>= 1) { if (t < st) red[t] += red[t + st]; __syncthreads(); }
  const float mean = red[0] * (1.0f / H_);
  __syncthreads();
  const float d = s - mean;
  red[t] = d * d; __syncthreads();
  for (int st = 128; st > 0; st >>= 1) { if (t < st) red[t] += red[t + st]; __syncthreads(); }
  const float rstd = rsqrtf(red[0] * (1.0f / H_) + 1e-5f);
  const float v = fmaxf((s - mean) * rstd * ls[t] + lb[t], 0.0f);
  h2[b * H_ + t]  = v;
  h2h[b * H_ + t] = (f16)v;
}

// ---------------- K3: logits = h2 @ gW3 + gb3 --------------------------------
// block = one 16-col N tile x all 128 rows; 8 waves, each owns a 16x16 C tile.
// Preferred path: TDM loads the whole 256x16 f32 B panel to LDS once, then
// fp32 WMMA (16x16x4) accumulates at full precision (GEMM is memory-bound, so
// the f32 matrix rate is free). Fallback: f16 WMMA with VALU-staged B tile.
__global__ __launch_bounds__(256)
void logits_gemm(const float* __restrict__ h2, const f16* __restrict__ h2h,
                 const float* __restrict__ gW3, const float* __restrict__ gb3,
                 float* __restrict__ logits) {
  const int n0 = blockIdx.x * 16;
  const int t = threadIdx.x;
  const int wave = t >> 5, lane = t & 31;
  const int nl = lane & 15;
  const int m  = wave * 16 + nl;            // A-matrix row for this lane
  v8f c = {};
#if USE_F32_WMMA_TDM
  typedef unsigned int u32x4 __attribute__((ext_vector_type(4)));
  typedef int          i32x8 __attribute__((ext_vector_type(8)));
  typedef int          i32x4 __attribute__((ext_vector_type(4)));
  __shared__ alignas(16) float ldsB[H_ * 16];   // [k][n] 256x16 f32 = 16 KB
  if (t < 32) {  // wave 0 issues one TDM descriptor for the whole B panel
    const unsigned lds_off = (unsigned)(size_t)(&ldsB[0]);      // LDS aperture: low 32 bits = LDS offset
    const unsigned long long ga = (unsigned long long)(const void*)(gW3 + n0);
    u32x4 g0;
    g0[0] = 1u;                                  // count=1, no gather
    g0[1] = lds_off;                             // lds_addr
    g0[2] = (unsigned)ga;                        // global_addr[31:0]
    g0[3] = (unsigned)(ga >> 32) | (2u << 30);   // global_addr[56:32] | type=2
    i32x8 g1;
    g1[0] = (2 << 16);                           // data_size=4B; mask/pad/iter=0
    g1[1] = (int)((ND_ & 0xFFFFu) << 16);        // tensor_dim0[15:0]
    g1[2] = (int)((ND_ >> 16) | ((H_ & 0xFFFFu) << 16)); // dim0[31:16] | dim1[15:0]
    g1[3] = (int)((16u) << 16);                  // dim1[31:16]=0 | tile_dim0=16
    g1[4] = (int)(H_ & 0xFFFFu);                 // tile_dim1=256, tile_dim2=0
    g1[5] = (int)ND_;                            // tensor_dim0_stride[31:0]
    g1[6] = 0;                                   // stride0 hi | stride1 lo
    g1[7] = 0;                                   // stride1 hi
    i32x4 gz;  gz[0] = 0; gz[1] = 0; gz[2] = 0; gz[3] = 0;
    i32x8 gz8; gz8[0] = 0; gz8[1] = 0; gz8[2] = 0; gz8[3] = 0;
    gz8[4] = 0; gz8[5] = 0; gz8[6] = 0; gz8[7] = 0;
    __builtin_amdgcn_tensor_load_to_lds(g0, g1, gz, gz, gz8, 0);  // 6-arg form
    __builtin_amdgcn_s_wait_tensorcnt(0);
  }
  __syncthreads();
  // fp32 WMMA: A 16x4 layout -> lanes 0-15 hold K={0,1}, lanes 16-31 K={2,3}
  const float* arow = h2 + m * H_;
  const int ksel = (lane < 16) ? 0 : 2;
  #pragma unroll 8
  for (int kc = 0; kc < H_; kc += 4) {
    const int k0 = kc + ksel;
    const v2f a = *(const v2f*)(arow + k0);                  // aligned b64 load
    v2f bfrag;
    bfrag[0] = ldsB[k0 * 16 + nl];
    bfrag[1] = ldsB[(k0 + 1) * 16 + nl];
    c = __builtin_amdgcn_wmma_f32_16x16x4_f32(false, a, false, bfrag,
                                              (short)0, c, false, false);
  }
#else
  __shared__ alignas(16) f16 ldsB[16 * 32];   // transposed tile: [n][k]
  const int kb = (lane < 16) ? 0 : 8;
  for (int kc = 0; kc < H_; kc += 32) {
    __syncthreads();
    for (int e = t; e < 512; e += 256) {    // stage + convert B tile (32x16 f32)
      const int k = e >> 4, n = e & 15;
      ldsB[n * 32 + k] = (f16)gW3[(size_t)(kc + k) * ND_ + n0 + n];
      if (kc + 32 < H_)
        __builtin_prefetch(&gW3[(size_t)(kc + 32 + k) * ND_ + n0 + n], 0, 0);
    }
    __syncthreads();
    const f16* arow = h2h + m * H_ + kc + kb;
    const v8h a0 = *(const v8h*)(arow);
    const v8h a1 = *(const v8h*)(arow + 16);
    const v16h A = __builtin_shufflevector(a0, a1, 0,1,2,3,4,5,6,7,8,9,10,11,12,13,14,15);
    const f16* brow = ldsB + nl * 32 + kb;
    const v8h b0 = *(const v8h*)(brow);
    const v8h b1 = *(const v8h*)(brow + 16);
    const v16h Bf = __builtin_shufflevector(b0, b1, 0,1,2,3,4,5,6,7,8,9,10,11,12,13,14,15);
    c = __builtin_amdgcn_wmma_f32_16x16x32_f16(false, A, false, Bf, (short)0, c, false, false);
  }
#endif
  const int col = n0 + nl;
  const float bias = gb3[col];
  const int rbase = wave * 16 + ((lane < 16) ? 0 : 8);
  #pragma unroll
  for (int i = 0; i < 8; ++i)
    logits[(size_t)(rbase + i) * ND_ + col] = c[i] + bias;
}

// ---------------- K4: softmax((logits+g)/tau) + top-5 indices -------------
__global__ __launch_bounds__(256)
void softmax_topk(const float* __restrict__ logits, const float* __restrict__ gumbel,
                  float* __restrict__ probs, int* __restrict__ idxOut) {
  __shared__ float red[256];
  __shared__ int   redi[256];
  const int bm = blockIdx.x, t = threadIdx.x;
  const size_t base = (size_t)bm * D_;
  float z[4];
  #pragma unroll
  for (int j = 0; j < 4; ++j) {
    const int d = t + j * 256;
    z[j] = (logits[base + d] + gumbel[base + d]) * 2.0f;   // 1/TAU = 2
  }
  float mx = fmaxf(fmaxf(z[0], z[1]), fmaxf(z[2], z[3]));
  red[t] = mx; __syncthreads();
  for (int s = 128; s > 0; s >>= 1) { if (t < s) red[t] = fmaxf(red[t], red[t + s]); __syncthreads(); }
  mx = red[0]; __syncthreads();
  float e[4]; float lsum = 0.f;
  #pragma unroll
  for (int j = 0; j < 4; ++j) { e[j] = __expf(z[j] - mx); lsum += e[j]; }
  red[t] = lsum; __syncthreads();
  for (int s = 128; s > 0; s >>= 1) { if (t < s) red[t] += red[t + s]; __syncthreads(); }
  const float inv = 1.0f / red[0];
  __syncthreads();
  #pragma unroll
  for (int j = 0; j < 4; ++j) probs[base + t + j * 256] = e[j] * inv;
  // top-K on z (softmax is monotone -> same ordering as probs)
  for (int r = 0; r < K_; ++r) {
    float lv = NEG_INF; int li = 0;
    #pragma unroll
    for (int j = 0; j < 4; ++j) if (z[j] > lv) { lv = z[j]; li = t + j * 256; }
    red[t] = lv; redi[t] = li;
    __syncthreads();
    for (int s = 128; s > 0; s >>= 1) {
      if (t < s) {
        const float v2 = red[t + s]; const int i2 = redi[t + s];
        if (v2 > red[t] || (v2 == red[t] && i2 < redi[t])) { red[t] = v2; redi[t] = i2; }
      }
      __syncthreads();
    }
    const int dsel = redi[0];
    __syncthreads();
    if (t == 0) idxOut[bm * K_ + r] = dsel;
    if ((dsel & 255) == t) z[dsel >> 8] = NEG_INF;  // mask winner locally
  }
}

// ---------------- K5: scorer + synergy heads (sparse gather + WMMA) -------
// block = 16 (b,m) rows; flat@W1 collapses to 5 gathered rows of W1.
__global__ __launch_bounds__(256)
void scorer_synergy(const float* __restrict__ x, const int* __restrict__ idx,
                    const float* __restrict__ sW1, const float* __restrict__ sb1,
                    const float* __restrict__ sW2, const float* __restrict__ sb2,
                    const float* __restrict__ sW3, const float* __restrict__ sb3,
                    const float* __restrict__ yW1, const float* __restrict__ yb1,
                    const float* __restrict__ yW2, const float* __restrict__ yb2,
                    float* __restrict__ scores, float* __restrict__ syn) {
  __shared__ int   sIdx[16][K_];
  __shared__ float sVal[16][K_];
  __shared__ alignas(16) f16 hA[16][H_];   // scorer hidden-1, f16 for WMMA
  __shared__ float h2t[16][128];
  __shared__ float predS[16][16];
  __shared__ float predY[16][16];
  const int t = threadIdx.x;
  const int baseRow = blockIdx.x * 16;
  if (t < 16 * K_) {
    const int r = t / K_, j = t % K_;
    const int bm = baseRow + r;
    const int d = idx[bm * K_ + j];
    sIdx[r][j] = d;
    sVal[r][j] = x[(bm / M_) * D_ + d];
  }
  __syncthreads();
  {
    const int r = t >> 4, c0 = t & 15;
    float yp = 0.f;
    #pragma unroll
    for (int i = 0; i < 16; ++i) {
      const int c = c0 + 16 * i;
      float a = sb1[c], ay = yb1[c];
      #pragma unroll
      for (int j = 0; j < K_; ++j) {
        const int row = j * D_ + sIdx[r][j];
        const float v = sVal[r][j];
        a  = fmaf(v, sW1[row * H_ + c], a);
        ay = fmaf(v, yW1[row * H_ + c], ay);
      }
      hA[r][c] = (f16)fmaxf(a, 0.f);
      yp = fmaf(fmaxf(ay, 0.f), yW2[c], yp);   // synergy hidden -> dot(yW2)
    }
    predY[r][c0] = yp;
  }
  __syncthreads();
  // stage-2: [16,256] @ sW2[256,128] via WMMA; wave w owns cols [16w,16w+16)
  const int wave = t >> 5, lane = t & 31;
  const int nl = lane & 15;
  const int kb = (lane < 16) ? 0 : 8;
  const int ncol = wave * 16 + nl;
  v8f c2 = {};
  for (int kc = 0; kc < H_; kc += 32) {
    const f16* arow = &hA[nl][kc + kb];
    const v8h a0 = *(const v8h*)(arow);
    const v8h a1 = *(const v8h*)(arow + 16);
    const v16h A = __builtin_shufflevector(a0, a1, 0,1,2,3,4,5,6,7,8,9,10,11,12,13,14,15);
    v16h Bf;
    #pragma unroll
    for (int i = 0; i < 16; ++i) {
      const int k = kc + kb + ((i < 8) ? 0 : 16) + (i & 7);
      Bf[i] = (f16)sW2[k * 128 + ncol];
    }
    c2 = __builtin_amdgcn_wmma_f32_16x16x32_f16(false, A, false, Bf, (short)0, c2, false, false);
  }
  const int rb = (lane < 16) ? 0 : 8;
  #pragma unroll
  for (int i = 0; i < 8; ++i) h2t[rb + i][ncol] = fmaxf(c2[i], 0.f);
  __syncthreads();
  {
    const int g = t & 15, rr = t >> 4;   // 16 threads per row, 8 cols each
    float sp = 0.f;
    #pragma unroll
    for (int q = 0; q < 8; ++q) sp = fmaf(h2t[rr][g * 8 + q], sW3[g * 8 + q], sp);
    predS[rr][g] = sp;
  }
  __syncthreads();
  if (t < 16) {
    float ss = sb3[0], ys = yb2[0];
    #pragma unroll
    for (int q = 0; q < 16; ++q) { ss += predS[t][q]; ys += predY[t][q]; }
    scores[baseRow + t] = 1.0f / (1.0f + __expf(-ss));
    syn[baseRow + t]    = tanhf(ys);
  }
}

extern "C" void kernel_launch(void* const* d_in, const int* in_sizes, int n_in,
                              void* d_out, int out_size, void* d_ws, size_t ws_size,
                              hipStream_t stream) {
  const float* x      = (const float*)d_in[0];
  const float* gumbel = (const float*)d_in[1];
  const float* gW1  = (const float*)d_in[2];  const float* gb1  = (const float*)d_in[3];
  const float* ln1s = (const float*)d_in[4];  const float* ln1b = (const float*)d_in[5];
  const float* gW2  = (const float*)d_in[6];  const float* gb2  = (const float*)d_in[7];
  const float* ln2s = (const float*)d_in[8];  const float* ln2b = (const float*)d_in[9];
  const float* gW3  = (const float*)d_in[10]; const float* gb3  = (const float*)d_in[11];
  const float* sW1  = (const float*)d_in[12]; const float* sb1  = (const float*)d_in[13];
  const float* sW2  = (const float*)d_in[14]; const float* sb2  = (const float*)d_in[15];
  const float* sW3  = (const float*)d_in[16]; const float* sb3  = (const float*)d_in[17];
  const float* yW1  = (const float*)d_in[18]; const float* yb1  = (const float*)d_in[19];
  const float* yW2  = (const float*)d_in[20]; const float* yb2  = (const float*)d_in[21];

  float* out    = (float*)d_out;          // (probs, scores, synergies, logits)
  float* probs  = out;
  float* scores = out + (size_t)B_ * M_ * D_;
  float* synv   = scores + B_ * M_;
  float* logits = synv + B_ * M_;

  char* ws = (char*)d_ws;
  float* h1     = (float*)ws;                                     // 128*512*4 = 256 KB
  float* h2     = (float*)(ws + (size_t)B_ * H2_ * 4);            // 128*256*4 = 128 KB
  f16*   h2h    = (f16*)(ws + (size_t)B_ * H2_ * 4
                            + (size_t)B_ * H_ * 4);               // 128*256*2 =  64 KB
  int*   idxbuf = (int*)(ws + (size_t)B_ * H2_ * 4
                            + (size_t)B_ * H_ * 4
                            + (size_t)B_ * H_ * 2);               // 12800*5*4 = 256 KB

  hipLaunchKernelGGL(gen_layer1,   dim3(B_),          dim3(256), 0, stream,
                     x, gW1, gb1, ln1s, ln1b, h1);
  hipLaunchKernelGGL(gen_layer2,   dim3(B_),          dim3(256), 0, stream,
                     h1, gW2, gb2, ln2s, ln2b, h2, h2h);
  hipLaunchKernelGGL(logits_gemm,  dim3(ND_ / 16),    dim3(256), 0, stream,
                     h2, h2h, gW3, gb3, logits);
  hipLaunchKernelGGL(softmax_topk, dim3(B_ * M_),     dim3(256), 0, stream,
                     logits, gumbel, probs, idxbuf);
  hipLaunchKernelGGL(scorer_synergy, dim3(B_ * M_ / 16), dim3(256), 0, stream,
                     x, idxbuf, sW1, sb1, sW2, sb2, sW3, sb3,
                     yW1, yb1, yW2, yb2, scores, synv);
}
